// SpacingRNN_9466107920519
// MI455X (gfx1250) — matrix operand order repaired and checked by reading, not compiled, not run on hardware
//
#include <hip/hip_runtime.h>
#include <hip/hip_bf16.h>

// ---------------------------------------------------------------------------
// SpacingRNN persistent kernel for MI455X (gfx1250, wave32, WMMA).
//
// Shapes: B=64, D=512, H=512, L_IN=1000, length=800, gates N=2048, K=1024.
// Design: single persistent kernel, 64 WGs x 256 thr, grid barriers via L2
// atomics, bf16 WMMA (f32 acc) for the recurrent GEMM with i/f/g/o fused
// in-register, attention kept in LDS per batch row.
// ---------------------------------------------------------------------------

typedef __attribute__((ext_vector_type(16))) __bf16 v16bf;
typedef __attribute__((ext_vector_type(8)))  float  v8f;

#define NBLK   64
#define NTHR   256
#define BATCH_ 64
#define DIM    512
#define HID    512
#define L_IN_  1000
#define KTOT   1024          // D + H
#define WPACK_DW (128*32*32*8)   // ntile * kc * lane * vgpr

union Frag {
  uint4    u4[2];
  unsigned u[8];
  v16bf    bf;
};

__device__ __forceinline__ unsigned f2bf(float f) {
  unsigned u = __float_as_uint(f);
  return (u + 0x7fffu + ((u >> 16) & 1u)) >> 16;   // RNE
}
__device__ __forceinline__ unsigned pack_bf2(float lo, float hi) {
  return f2bf(lo) | (f2bf(hi) << 16);
}

// Dword index inside the packed A-matrix buffer for element (mtile, m, k)
// (k even; the dword also holds k+1). Matches the 16-bit 16x32 A layout:
// lanes 0-15: K 0-7 in v0-3, K 16-23 in v4-7; lanes 16-31: K 8-15 / 24-31.
__device__ __forceinline__ int a_idx(int mtile, int m, int k) {
  int kc = k >> 5;
  int kk = k & 31;
  int kb = kk & 15;
  int g  = kb >> 3;
  int kr = kb & 7;
  int v  = ((kk >> 4) << 2) + (kr >> 1);
  int lane = m + (g << 4);
  return ((mtile * 32 + kc) * 32 + lane) * 8 + v;
}

__device__ __forceinline__ float sigmoidf_(float x) {
  return 1.0f / (1.0f + __expf(-x));
}

// ---------------------------------------------------------------------------
// Setup kernel 1: pack W = [W_ih ; W_hh] (K=1024, N=2048) as bf16 B-fragments.
// B-frag layout (16-bit, 32x16): lane n in 0-15 -> cols; lanes 0-15 hold
// K 0-7 in v0-3, K 16-23 in v4-7; lanes 16-31 hold K 8-15 / 24-31.
// ---------------------------------------------------------------------------
__global__ void pack_weights(const float* __restrict__ Wih,
                             const float* __restrict__ Whh,
                             unsigned* __restrict__ Wpack) {
  int idx   = blockIdx.x * blockDim.x + threadIdx.x;   // < WPACK_DW
  int v     = idx & 7;
  int lane  = (idx >> 3) & 31;
  int kc    = (idx >> 8) & 31;
  int ntile = idx >> 13;
  int n = lane & 15;
  int g = lane >> 4;
  int j  = ntile * 16 + n;          // gate output column (0..2047)
  int k0 = kc * 32 + (v >> 2) * 16 + g * 8 + 2 * (v & 3);  // even
  float w0 = (k0 < DIM) ? Wih[j * DIM + k0] : Whh[j * HID + (k0 - DIM)];
  float w1 = (k0 + 1 < DIM) ? Wih[j * DIM + k0 + 1]
                            : Whh[j * HID + (k0 + 1 - DIM)];
  Wpack[idx] = pack_bf2(w0, w1);
}

// ---------------------------------------------------------------------------
// Setup kernel 2: zero c-state and packed activations (h0 = 0), fold biases,
// reset barrier counters. Grid = 128 x 256 = 32768 threads.
// ---------------------------------------------------------------------------
__global__ void init_state(float* __restrict__ cst, unsigned* __restrict__ actA,
                           float* __restrict__ bsum,
                           const float* __restrict__ bih,
                           const float* __restrict__ bhh,
                           unsigned* __restrict__ sync) {
  int tid = blockIdx.x * blockDim.x + threadIdx.x;
  cst[tid]  = 0.0f;
  actA[tid] = 0u;
  if (tid < 2048) bsum[tid] = bih[tid] + bhh[tid];
  if (tid < 8)    sync[tid] = 0u;
}

// ---------------------------------------------------------------------------
// Device-wide barrier (sense via generation counter in L2).
// ---------------------------------------------------------------------------
__device__ __forceinline__ void grid_sync(unsigned* cnt, unsigned* gen) {
  __threadfence();
  __syncthreads();
  if (threadIdx.x == 0) {
    unsigned g = __hip_atomic_load(gen, __ATOMIC_ACQUIRE, __HIP_MEMORY_SCOPE_AGENT);
    unsigned a = __hip_atomic_fetch_add(cnt, 1u, __ATOMIC_ACQ_REL, __HIP_MEMORY_SCOPE_AGENT);
    if (a == (unsigned)(gridDim.x - 1)) {
      __hip_atomic_store(cnt, 0u, __ATOMIC_RELAXED, __HIP_MEMORY_SCOPE_AGENT);
      __hip_atomic_fetch_add(gen, 1u, __ATOMIC_RELEASE, __HIP_MEMORY_SCOPE_AGENT);
    } else {
      while (__hip_atomic_load(gen, __ATOMIC_ACQUIRE, __HIP_MEMORY_SCOPE_AGENT) == g) {
        __builtin_amdgcn_s_sleep(1);
      }
    }
  }
  __syncthreads();
}

// ---------------------------------------------------------------------------
// Persistent kernel: 800 recurrent steps, 2 grid barriers per step.
// ---------------------------------------------------------------------------
__global__ void __launch_bounds__(NTHR)
spacing_rnn(const float* __restrict__ input,     // (L_IN, B, D) f32
            const int*   __restrict__ plen,
            const unsigned* __restrict__ Wpack,
            float*    __restrict__ cst,          // (B, H) f32 cell state
            unsigned* __restrict__ actA,         // packed bf16 A fragments
            const float* __restrict__ bsum,      // b_ih + b_hh (2048)
            const float* __restrict__ Wsp,       // (2, H)
            const float* __restrict__ bsp,       // (2,)
            float* __restrict__ outs,            // (length, B, H)
            unsigned* __restrict__ sync,
            int max_t) {
  __shared__ float s_attn[L_IN_];
  __shared__ float s_red[NTHR];
  __shared__ float s_bc[2];

  const int tx   = threadIdx.x;
  const int b    = blockIdx.x;          // batch row owned for phases A & D
  const int lane = tx & 31;
  const int w    = tx >> 5;

  for (int l = tx; l < L_IN_; l += NTHR) s_attn[l] = (l == 0) ? 1.0f : 0.0f;
  __syncthreads();

  int len = *plen;
  if (len > max_t) len = max_t;

  // --- WMMA wave task: 128 tasks = 32 column-groups x 4 batch tiles.
  // Waves 0,1 of every WG are active; each owns h-cols [ng*16, ng*16+16)
  // across all four gates, for batch rows [mtile*16, mtile*16+16).
  const bool bc_active = (w < 2);
  const int  task  = blockIdx.x * 2 + w;     // 0..127
  const int  ng    = task & 31;
  const int  mtile = task >> 5;
  const int  ncol  = lane & 15;
  const int  hcol  = ng * 16 + ncol;
  const int  brow0 = mtile * 16 + (lane >> 4) * 8;   // + r (0..7)

  const uint4* A4 = reinterpret_cast<const uint4*>(actA);
  const uint4* W4 = reinterpret_cast<const uint4*>(Wpack);

  float bias_i = 0.f, bias_f = 0.f, bias_g = 0.f, bias_o = 0.f;
  if (bc_active) {
    bias_i = bsum[hcol];
    bias_f = bsum[512  + hcol];
    bias_g = bsum[1024 + hcol];
    bias_o = bsum[1536 + hcol];
  }

  const int mA  = b & 15;      // phase A/D activation-pack coordinates
  const int mtA = b >> 4;

  for (int t = 0; t < len; ++t) {
    // ---------------- Phase A: time_in[b, 2tx..2tx+1] = x[b,:,:] @ attn[b,:]
    {
      float acc0 = 0.f, acc1 = 0.f;
      const float2* xin =
          reinterpret_cast<const float2*>(input) + (size_t)b * 256 + tx;
      #pragma unroll 4
      for (int l = 0; l < L_IN_; ++l) {
        float  a = s_attn[l];
        float2 x = xin[(size_t)l * (BATCH_ * 256)];
        acc0 = fmaf(x.x, a, acc0);
        acc1 = fmaf(x.y, a, acc1);
      }
      actA[a_idx(mtA, mA, 2 * tx)] = pack_bf2(acc0, acc1);
    }
    grid_sync(sync, sync + 1);

    // ---------------- Phase B+C: gates via WMMA, LSTM fused in-register
    if (bc_active) {
      v8f ci = {}, cf = {}, cg = {}, co = {};
      for (int kc = 0; kc < 32; ++kc) {
        int abase = ((mtile * 32 + kc) * 32 + lane) * 2;   // uint4 index
        Frag fa; fa.u4[0] = A4[abase]; fa.u4[1] = A4[abase + 1];
        int wi = (((ng     ) * 32 + kc) * 32 + lane) * 2;
        int wf = (((32 + ng) * 32 + kc) * 32 + lane) * 2;
        int wg = (((64 + ng) * 32 + kc) * 32 + lane) * 2;
        int wo = (((96 + ng) * 32 + kc) * 32 + lane) * 2;
        Frag f0; f0.u4[0] = W4[wi]; f0.u4[1] = W4[wi + 1];
        Frag f1; f1.u4[0] = W4[wf]; f1.u4[1] = W4[wf + 1];
        Frag f2; f2.u4[0] = W4[wg]; f2.u4[1] = W4[wg + 1];
        Frag f3; f3.u4[0] = W4[wo]; f3.u4[1] = W4[wo + 1];
        ci = __builtin_amdgcn_wmma_f32_16x16x32_bf16(false, fa.bf, false, f0.bf,
                                                     (short)0, ci, false, false);
        cf = __builtin_amdgcn_wmma_f32_16x16x32_bf16(false, fa.bf, false, f1.bf,
                                                     (short)0, cf, false, false);
        cg = __builtin_amdgcn_wmma_f32_16x16x32_bf16(false, fa.bf, false, f2.bf,
                                                     (short)0, cg, false, false);
        co = __builtin_amdgcn_wmma_f32_16x16x32_bf16(false, fa.bf, false, f3.bf,
                                                     (short)0, co, false, false);
      }
      #pragma unroll
      for (int r = 0; r < 8; ++r) {
        int   bb = brow0 + r;
        float iv = sigmoidf_(ci[r] + bias_i);
        float fv = sigmoidf_(cf[r] + bias_f);
        float gv = tanhf(cg[r] + bias_g);
        float ov = sigmoidf_(co[r] + bias_o);
        float c  = cst[bb * HID + hcol];
        float c2 = fv * c + iv * gv;
        float h2 = ov * tanhf(c2);
        cst[bb * HID + hcol] = c2;
        outs[((size_t)t * BATCH_ + bb) * HID + hcol] = h2;
      }
    }
    grid_sync(sync, sync + 1);

    // ---------------- Phase D: sp head + attention update (per-b WG, LDS)
    {
      const float* hrow = outs + ((size_t)t * BATCH_ + b) * HID;
      float h0 = hrow[2 * tx], h1 = hrow[2 * tx + 1];
      // repack h for next step's A fragments (K = 512..1023)
      actA[a_idx(mtA, mA, DIM + 2 * tx)] = pack_bf2(h0, h1);
      float p0 = h0 * Wsp[2 * tx]       + h1 * Wsp[2 * tx + 1];
      float p1 = h0 * Wsp[HID + 2 * tx] + h1 * Wsp[HID + 2 * tx + 1];

      s_red[tx] = p0; __syncthreads();
      for (int s = 128; s > 0; s >>= 1) {
        if (tx < s) s_red[tx] += s_red[tx + s];
        __syncthreads();
      }
      if (tx == 0) s_bc[0] = s_red[0] + bsp[0];
      __syncthreads();
      s_red[tx] = p1; __syncthreads();
      for (int s = 128; s > 0; s >>= 1) {
        if (tx < s) s_red[tx] += s_red[tx + s];
        __syncthreads();
      }
      if (tx == 0) s_bc[1] = s_red[0] + bsp[1];
      __syncthreads();

      float shift = sigmoidf_(s_bc[0]);
      float sharp = fmaxf(s_bc[1], 0.f) + 1.0f;

      float pw[4]; float psum = 0.f;
      #pragma unroll
      for (int q = 0; q < 4; ++q) {
        int l = tx + q * NTHR;
        float val = 0.f;
        if (l < L_IN_) {
          float a  = s_attn[l];
          float ap = (l > 0) ? s_attn[l - 1] : 0.f;
          float a2 = (1.f - shift) * a + shift * ap;
          val = (a2 > 0.f) ? __expf(sharp * __logf(a2)) : 0.f;
        }
        pw[q] = val;
        psum += val;
      }
      __syncthreads();              // all s_attn reads complete
      s_red[tx] = psum; __syncthreads();
      for (int s = 128; s > 0; s >>= 1) {
        if (tx < s) s_red[tx] += s_red[tx + s];
        __syncthreads();
      }
      float inv = 1.0f / s_red[0];
      #pragma unroll
      for (int q = 0; q < 4; ++q) {
        int l = tx + q * NTHR;
        if (l < L_IN_) s_attn[l] = pw[q] * inv;
      }
      __syncthreads();
    }
    // No barrier needed: next phase A touches only this WG's LDS + its own
    // actA rows; the next step's first grid_sync orders everything else.
  }
}

// ---------------------------------------------------------------------------
extern "C" void kernel_launch(void* const* d_in, const int* in_sizes, int n_in,
                              void* d_out, int out_size, void* d_ws,
                              size_t ws_size, hipStream_t stream) {
  const float* input = (const float*)d_in[0];   // (1000, 64, 512)
  const int*   plen  = (const int*)  d_in[1];
  const float* Wih   = (const float*)d_in[2];   // (2048, 512)
  const float* Whh   = (const float*)d_in[3];   // (2048, 512)
  const float* bih   = (const float*)d_in[4];
  const float* bhh   = (const float*)d_in[5];
  const float* Wsp   = (const float*)d_in[6];   // (2, 512)
  const float* bsp   = (const float*)d_in[7];
  float* outs = (float*)d_out;

  // Workspace layout (dwords): Wpack | cst | actA | bsum | sync  (~4.3 MB)
  unsigned* Wpack = (unsigned*)d_ws;
  float*    cst   = (float*)(Wpack + WPACK_DW);
  unsigned* actA  = (unsigned*)(cst + BATCH_ * HID);
  float*    bsum  = (float*)(actA + 4 * 32 * 32 * 8);
  unsigned* sync  = (unsigned*)(bsum + 2048);

  pack_weights<<<WPACK_DW / NTHR, NTHR, 0, stream>>>(Wih, Whh, Wpack);
  init_state<<<(BATCH_ * HID) / NTHR, NTHR, 0, stream>>>(cst, actA, bsum, bih,
                                                         bhh, sync);
  int max_t = out_size / (BATCH_ * HID);
  spacing_rnn<<<NBLK, NTHR, 0, stream>>>(input, plen, Wpack, cst, actA, bsum,
                                         Wsp, bsp, outs, sync, max_t);
}
